// TPSTransformer_69956427317826
// MI455X (gfx1250) — matrix operand order repaired and checked by reading, not compiled, not run on hardware
//
#include <hip/hip_runtime.h>
#include <hip/hip_bf16.h>
#include <math.h>

typedef __attribute__((ext_vector_type(16))) _Float16 v16h;
typedef __attribute__((ext_vector_type(8)))  float    v8f;

// f1c: conv1 output, channel-last with 1-pixel zero halo:
//   f1c[((b*130 + (y+1))*130 + (x+1))*32 + ic],  y,x in [0,128)
#define F1C_BATCH ((size_t)130 * 130 * 32)

// ---------------------------------------------------------------------------
// Kernel 1: conv1 (1->32, 3x3, SAME) + bias + maxpool2 + relu -> f16
// channel-last store (oc fastest across lanes -> coalesced b16 stores;
// all 32 lanes of a cluster share the same input pixels -> broadcast loads)
// ---------------------------------------------------------------------------
__global__ __launch_bounds__(256)
void k_conv1(const float* __restrict__ x, const float* __restrict__ w,
             const float* __restrict__ bias, _Float16* __restrict__ f1c) {
    int tid = blockIdx.x * blockDim.x + threadIdx.x;   // 32*128*128*32
    int oc = tid & 31;
    int px = (tid >> 5) & 127;
    int py = (tid >> 12) & 127;
    int b  = tid >> 19;

    float wv[9];
#pragma unroll
    for (int i = 0; i < 9; ++i) wv[i] = w[oc * 9 + i];
    float bv = bias[oc];

    const float* xb = x + (size_t)b * 256 * 256;
    float m = -INFINITY;
#pragma unroll
    for (int sy = 0; sy < 2; ++sy) {
#pragma unroll
        for (int sx = 0; sx < 2; ++sx) {
            int yc = py * 2 + sy, xc = px * 2 + sx;
            float acc = bv;
#pragma unroll
            for (int ky = 0; ky < 3; ++ky) {
#pragma unroll
                for (int kx = 0; kx < 3; ++kx) {
                    int yy = yc + ky - 1, xx = xc + kx - 1;
                    bool ok = (yy >= 0) & (yy < 256) & (xx >= 0) & (xx < 256);
                    int yyc = min(max(yy, 0), 255), xxc = min(max(xx, 0), 255);
                    float v = xb[yyc * 256 + xxc];
                    acc = fmaf(wv[ky * 3 + kx], ok ? v : 0.0f, acc);
                }
            }
            m = fmaxf(m, acc);
        }
    }
    m = fmaxf(m, 0.0f);
    f1c[((size_t)b * 130 + (py + 1)) * 130 * 32 + (size_t)(px + 1) * 32 + oc] = (_Float16)m;
}

// ---------------------------------------------------------------------------
// Kernel 1b: zero the 1-pixel halo ring of f1c (516 pixels/batch * 32 ch)
// ---------------------------------------------------------------------------
__global__ void k_halo(_Float16* __restrict__ f1c) {
    int tid = blockIdx.x * blockDim.x + threadIdx.x;   // 32 * 516 * 32
    if (tid >= 32 * 516 * 32) return;
    int ic = tid & 31;
    int p  = (tid >> 5) % 516;
    int b  = tid / (516 * 32);
    int y, x;
    if (p < 130)      { y = 0;       x = p; }
    else if (p < 260) { y = 129;     x = p - 130; }
    else if (p < 388) { y = p - 259; x = 0; }       // y = 1..128
    else              { y = p - 387; x = 129; }     // y = 1..128
    f1c[((size_t)b * 130 + y) * 130 * 32 + (size_t)x * 32 + ic] = (_Float16)0.0f;
}

// ---------------------------------------------------------------------------
// Kernel 2: pack conv2 weights (OIHW f32) into per-lane WMMA A-fragment
// order: w2p[((t*4 + mt)*32 + lane)*16 + j]
//   m = mt*16 + (lane&15); kb = (lane>=16)?16:0; k = kb + j
// ---------------------------------------------------------------------------
__global__ void k_w2pack(const float* __restrict__ w2, _Float16* __restrict__ w2p) {
    int tid = blockIdx.x * blockDim.x + threadIdx.x;   // 9*4*32*16 = 18432
    if (tid >= 9 * 4 * 32 * 16) return;
    int j  = tid & 15;
    int l  = (tid >> 4) & 31;
    int mt = (tid >> 9) & 3;
    int t  = tid >> 11;                // 0..8
    int m  = mt * 16 + (l & 15);
    int kb = (l >> 4) * 16;
    int k  = kb + j;
    w2p[tid] = (_Float16)w2[(m * 32 + k) * 9 + t];
}

// ---------------------------------------------------------------------------
// Kernel 3: conv2 (32->64, 3x3, SAME) WMMA f16 implicit GEMM,
// fused bias + maxpool2 + relu.  One wave per block; each wave owns a
// 2x8 pixel tile (N=16) x 64 output channels (4 M-tiles).
// A fragment: one contiguous v16h per lane (pre-packed, coalesced).
// B fragment: one contiguous v16h per lane (channel-last + halo, no selects).
// ---------------------------------------------------------------------------
__global__ __launch_bounds__(32)
void k_conv2_wmma(const _Float16* __restrict__ f1c, const _Float16* __restrict__ w2p,
                  const float* __restrict__ bias2, float* __restrict__ c2p) {
    __shared__ float tile[64][16];

    int l   = threadIdx.x;          // lane 0..31
    int id  = blockIdx.x;           // 32 batches * 1024 tiles
    int b   = id >> 10;
    int rem = id & 1023;
    int ty  = rem >> 4;             // 0..63 : pooled row (pixel row pair)
    int tx  = rem & 15;             // 0..15 : group of 8 pixel cols

    int n    = l & 15;              // B/N column this lane owns
    int y    = ty * 2 + (n >> 3);   // pixel row (0..127)
    int xpix = tx * 8 + (n & 7);    // pixel col (0..127)
    int kbB  = (l >> 4) * 16;       // B K-offset: contiguous k = 16h + s, same order as A pack

    const _Float16* fb = f1c + (size_t)b * F1C_BATCH;

    v8f acc0 = {}, acc1 = {}, acc2 = {}, acc3 = {};

#pragma unroll
    for (int t = 0; t < 9; ++t) {
        int dy = t / 3 - 1, dx = t % 3 - 1;
        int yy = y + dy + 1, xx = xpix + dx + 1;   // halo coords, always in-bounds
        v16h bm = *(const v16h*)(fb + ((size_t)yy * 130 + xx) * 32 + kbB);

        const _Float16* wp = w2p + ((size_t)t * 4 * 32) * 16 + (size_t)l * 16;
        v16h a0 = *(const v16h*)(wp);
        v16h a1 = *(const v16h*)(wp + 32 * 16);
        v16h a2 = *(const v16h*)(wp + 64 * 16);
        v16h a3 = *(const v16h*)(wp + 96 * 16);

        acc0 = __builtin_amdgcn_wmma_f32_16x16x32_f16(false, a0, false, bm, (short)0, acc0, false, false);
        acc1 = __builtin_amdgcn_wmma_f32_16x16x32_f16(false, a1, false, bm, (short)0, acc1, false, false);
        acc2 = __builtin_amdgcn_wmma_f32_16x16x32_f16(false, a2, false, bm, (short)0, acc2, false, false);
        acc3 = __builtin_amdgcn_wmma_f32_16x16x32_f16(false, a3, false, bm, (short)0, acc3, false, false);
    }

    // D layout: VGPR r -> row = r + 8*(l>=16), col = l&15
    int rowoff = (l >> 4) * 8;
    int col    = l & 15;
#pragma unroll
    for (int r = 0; r < 8; ++r) {
        tile[ 0 + rowoff + r][col] = acc0[r];
        tile[16 + rowoff + r][col] = acc1[r];
        tile[32 + rowoff + r][col] = acc2[r];
        tile[48 + rowoff + r][col] = acc3[r];
    }
    __syncthreads();

    // 2x2 maxpool (cols 0-7 = row y0, cols 8-15 = row y0+1) + bias + relu
#pragma unroll
    for (int g = 0; g < 2; ++g) {
        int oc = l + g * 32;
        float bv = bias2[oc];
#pragma unroll
        for (int c = 0; c < 4; ++c) {
            float v = fmaxf(fmaxf(tile[oc][c * 2], tile[oc][c * 2 + 1]),
                            fmaxf(tile[oc][8 + c * 2], tile[oc][8 + c * 2 + 1]));
            v = fmaxf(v + bv, 0.0f);
            c2p[(((size_t)b * 64 + oc) * 64 + ty) * 64 + tx * 4 + c] = v;
        }
    }
}

// ---------------------------------------------------------------------------
// Kernel 4: 16x16 average pool -> feature vector (b, c*16 + i*4 + j)
// ---------------------------------------------------------------------------
__global__ void k_avgpool(const float* __restrict__ c2p, float* __restrict__ feat) {
    int tid = blockIdx.x * blockDim.x + threadIdx.x;   // 32768
    int j = tid & 3;
    int i = (tid >> 2) & 3;
    int c = (tid >> 4) & 63;
    int b = tid >> 10;
    const float* p = c2p + (((size_t)b * 64 + c) * 64 + i * 16) * 64 + j * 16;
    float s = 0.0f;
    for (int yy = 0; yy < 16; ++yy)
#pragma unroll
        for (int xx = 0; xx < 16; ++xx) s += p[yy * 64 + xx];
    feat[tid] = s * (1.0f / 256.0f);
}

// ---------------------------------------------------------------------------
// Kernel 5: fc1 (1024 -> 256) + relu. one thread per (b, o)
// ---------------------------------------------------------------------------
__global__ void k_fc1(const float* __restrict__ feat, const float* __restrict__ w,
                      const float* __restrict__ bias, float* __restrict__ h) {
    int tid = blockIdx.x * blockDim.x + threadIdx.x;   // 8192
    int o = tid & 255;
    int b = tid >> 8;
    const float* f  = feat + (size_t)b * 1024;
    const float* wr = w + (size_t)o * 1024;
    float acc = bias[o];
    for (int i = 0; i < 1024; ++i) acc = fmaf(f[i], wr[i], acc);
    h[tid] = fmaxf(acc, 0.0f);
}

// ---------------------------------------------------------------------------
// Kernel 6: fc2 (256 -> 18) + build TPS system + 12x12 solve (per batch).
// one thread per batch image (32 threads total).
// ---------------------------------------------------------------------------
__global__ void k_fc2_solve(const float* __restrict__ h, const float* __restrict__ w,
                            const float* __restrict__ bias, float* __restrict__ wf) {
    int b = threadIdx.x;
    const float* hb = h + (size_t)b * 256;

    float theta[18];
    for (int o = 0; o < 18; ++o) {
        const float* wr = w + (size_t)o * 256;
        float acc = bias[o];
        for (int i = 0; i < 256; ++i) acc = fmaf(hb[i], wr[i], acc);
        theta[o] = acc;
    }

    float cpx[9], cpy[9];
#pragma unroll
    for (int i = 0; i < 9; ++i) {
        cpx[i] = (float)(i % 3) - 1.0f;
        cpy[i] = (float)(i / 3) - 1.0f;
    }

    float Lm[12][12];
    for (int i = 0; i < 12; ++i)
        for (int j = 0; j < 12; ++j) Lm[i][j] = 0.0f;
    for (int i = 0; i < 9; ++i)
        for (int j = 0; j < 9; ++j) {
            float dx = cpx[i] - cpx[j], dy = cpy[i] - cpy[j];
            float r2 = dx * dx + dy * dy;
            Lm[i][j] = (r2 > 0.0f) ? r2 * logf(r2) : 0.0f;
        }
    for (int i = 0; i < 9; ++i) {
        Lm[i][9] = 1.0f; Lm[i][10] = cpx[i]; Lm[i][11] = cpy[i];
        Lm[9][i] = 1.0f; Lm[10][i] = cpx[i]; Lm[11][i] = cpy[i];
    }

    float Y[12][2];
    for (int k = 0; k < 9; ++k) {
        Y[k][0] = cpx[k] + theta[2 * k];
        Y[k][1] = cpy[k] + theta[2 * k + 1];
    }
    for (int k = 9; k < 12; ++k) { Y[k][0] = 0.0f; Y[k][1] = 0.0f; }

    for (int col = 0; col < 12; ++col) {
        int piv = col;
        float best = fabsf(Lm[col][col]);
        for (int r = col + 1; r < 12; ++r) {
            float v = fabsf(Lm[r][col]);
            if (v > best) { best = v; piv = r; }
        }
        if (piv != col) {
            for (int j = col; j < 12; ++j) {
                float t = Lm[col][j]; Lm[col][j] = Lm[piv][j]; Lm[piv][j] = t;
            }
            for (int d = 0; d < 2; ++d) {
                float t = Y[col][d]; Y[col][d] = Y[piv][d]; Y[piv][d] = t;
            }
        }
        float inv = 1.0f / Lm[col][col];
        for (int r = col + 1; r < 12; ++r) {
            float f = Lm[r][col] * inv;
            for (int j = col; j < 12; ++j) Lm[r][j] -= f * Lm[col][j];
            Y[r][0] -= f * Y[col][0];
            Y[r][1] -= f * Y[col][1];
        }
    }
    float Wv[12][2];
    for (int r = 11; r >= 0; --r) {
        float s0 = Y[r][0], s1 = Y[r][1];
        for (int j = r + 1; j < 12; ++j) {
            s0 -= Lm[r][j] * Wv[j][0];
            s1 -= Lm[r][j] * Wv[j][1];
        }
        float inv = 1.0f / Lm[r][r];
        Wv[r][0] = s0 * inv;
        Wv[r][1] = s1 * inv;
    }
    for (int r = 0; r < 12; ++r) {
        wf[(b * 12 + r) * 2 + 0] = Wv[r][0];
        wf[(b * 12 + r) * 2 + 1] = Wv[r][1];
    }
}

// ---------------------------------------------------------------------------
// Kernel 7: TPS radial basis U[n][k] (batch-invariant, L2-resident)
// ---------------------------------------------------------------------------
__global__ void k_tpsU(float* __restrict__ U) {
    int n = blockIdx.x * blockDim.x + threadIdx.x;   // 65536
    float gx = (float)(n & 255) * (2.0f / 255.0f) - 1.0f;
    float gy = (float)(n >> 8) * (2.0f / 255.0f) - 1.0f;
#pragma unroll
    for (int k = 0; k < 9; ++k) {
        float cx = (float)(k % 3) - 1.0f, cy = (float)(k / 3) - 1.0f;
        float dx = gx - cx, dy = gy - cy;
        float r2 = dx * dx + dy * dy;
        U[(size_t)n * 9 + k] = (r2 > 0.0f) ? r2 * logf(r2) : 0.0f;
    }
}

// ---------------------------------------------------------------------------
// Kernel 8: fused warp-field evaluation + bilinear grid_sample (zero pad)
// ---------------------------------------------------------------------------
__global__ __launch_bounds__(256)
void k_warp_sample(const float* __restrict__ x, const float* __restrict__ wf,
                   const float* __restrict__ U, float* __restrict__ out) {
    int tid = blockIdx.x * blockDim.x + threadIdx.x;   // 32*65536
    int n = tid & 65535;
    int b = tid >> 16;
    const float* W = wf + (size_t)b * 24;

    float gx = (float)(n & 255) * (2.0f / 255.0f) - 1.0f;
    float gy = (float)(n >> 8) * (2.0f / 255.0f) - 1.0f;

    float sx = W[18] + W[20] * gx + W[22] * gy;   // A0 + A1*x + A2*y (d=0)
    float sy = W[19] + W[21] * gx + W[23] * gy;   // (d=1)

    const float* Un = U + (size_t)n * 9;
#pragma unroll
    for (int k = 0; k < 9; ++k) {
        float u = Un[k];
        sx = fmaf(u, W[k * 2 + 0], sx);
        sy = fmaf(u, W[k * 2 + 1], sy);
    }

    float fx = (sx + 1.0f) * 0.5f * 255.0f;
    float fy = (sy + 1.0f) * 0.5f * 255.0f;
    float x0f = floorf(fx), y0f = floorf(fy);
    float wx = fx - x0f, wy = fy - y0f;
    int x0 = (int)x0f, y0 = (int)y0f;

    const float* img = x + (size_t)b * 65536;
    auto fetch = [&](int ix, int iy) -> float {
        bool ok = (ix >= 0) & (ix < 256) & (iy >= 0) & (iy < 256);
        int ic = min(max(ix, 0), 255), iyc = min(max(iy, 0), 255);
        float v = img[iyc * 256 + ic];
        return ok ? v : 0.0f;
    };
    float v00 = fetch(x0, y0);
    float v10 = fetch(x0 + 1, y0);
    float v01 = fetch(x0, y0 + 1);
    float v11 = fetch(x0 + 1, y0 + 1);

    float r = v00 * (1.0f - wx) * (1.0f - wy) + v10 * wx * (1.0f - wy) +
              v01 * (1.0f - wx) * wy + v11 * wx * wy;
    out[tid] = r;
}

// ---------------------------------------------------------------------------
// Host launcher
// ---------------------------------------------------------------------------
extern "C" void kernel_launch(void* const* d_in, const int* in_sizes, int n_in,
                              void* d_out, int out_size, void* d_ws, size_t ws_size,
                              hipStream_t stream) {
    (void)in_sizes; (void)n_in; (void)out_size; (void)ws_size;

    const float* x       = (const float*)d_in[0];
    const float* conv1_w = (const float*)d_in[1];
    const float* conv1_b = (const float*)d_in[2];
    const float* conv2_w = (const float*)d_in[3];
    const float* conv2_b = (const float*)d_in[4];
    const float* fc1_w   = (const float*)d_in[5];
    const float* fc1_b   = (const float*)d_in[6];
    const float* fc2_w   = (const float*)d_in[7];
    const float* fc2_b   = (const float*)d_in[8];
    float* out = (float*)d_out;

    char* ws = (char*)d_ws;
    size_t off = 0;
    _Float16* f1c = (_Float16*)(ws + off); off += (size_t)32 * F1C_BATCH * 2;     // ~34.6 MB
    off = (off + 255) & ~(size_t)255;
    float*    c2p = (float*)(ws + off);    off += (size_t)32 * 64 * 64 * 64 * 4;  // 32 MB
    _Float16* w2p = (_Float16*)(ws + off); off += (size_t)9 * 4 * 32 * 16 * 2;    // 36 KB
    off = (off + 255) & ~(size_t)255;
    float*   feat = (float*)(ws + off);    off += (size_t)32 * 1024 * 4;
    float*   hbuf = (float*)(ws + off);    off += (size_t)32 * 256 * 4;
    float*     wf = (float*)(ws + off);    off += (size_t)32 * 12 * 2 * 4 + 256;
    float*      U = (float*)(ws + off);    off += (size_t)65536 * 9 * 4;

    k_conv1<<<65536, 256, 0, stream>>>(x, conv1_w, conv1_b, f1c);
    k_halo<<<(32 * 516 * 32 + 255) / 256, 256, 0, stream>>>(f1c);
    k_w2pack<<<(18432 + 255) / 256, 256, 0, stream>>>(conv2_w, w2p);
    k_conv2_wmma<<<32 * 1024, 32, 0, stream>>>(f1c, w2p, conv2_b, c2p);
    k_avgpool<<<32768 / 256, 256, 0, stream>>>(c2p, feat);
    k_fc1<<<8192 / 256, 256, 0, stream>>>(feat, fc1_w, fc1_b, hbuf);
    k_fc2_solve<<<1, 32, 0, stream>>>(hbuf, fc2_w, fc2_b, wf);
    k_tpsU<<<65536 / 256, 256, 0, stream>>>(U);
    k_warp_sample<<<(32 * 65536) / 256, 256, 0, stream>>>(x, wf, U, out);
}
